// AttentionBlock3D_15118284882329
// MI455X (gfx1250) — matrix operand order repaired and checked by reading, not compiled, not run on hardware
//
#include <hip/hip_runtime.h>

// ---------------------------------------------------------------------------
// AttentionBlock3D for MI455X (gfx1250, wave32, WMMA).
// C=256, N=4096, heads=8, hd=32, groups=32. All GEMMs via
// v_wmma_f32_16x16x32_bf16 (bf16 inputs, f32 accumulate).
// ---------------------------------------------------------------------------

typedef __bf16 bf16_t;
typedef __attribute__((ext_vector_type(16))) __bf16 v16bf;
typedef __attribute__((ext_vector_type(8)))  __bf16 v8bf;
typedef __attribute__((ext_vector_type(8)))  float  v8f;

#define CC      256
#define NN      4096
#define HEADS   8
#define HD      32
#define GROUPS  32
#define GSIZE   (8 * NN)   // elements per group

union V16U { v16bf v; v8bf h[2]; };

// A-operand (16x32 bf16, M x K) load, row-major source with leading dim ld.
// Lane L: m = L&15, b = L>>4 ; elems 0..15 = K[b*16 .. b*16+15].
__device__ __forceinline__ v16bf load_A16x32(const bf16_t* base, int ld, int lane) {
    int m = lane & 15, b = lane >> 4;
    const bf16_t* p = base + m * ld + b * 16;
    V16U u;
    u.h[0] = *(const v8bf*)(p);
    u.h[1] = *(const v8bf*)(p + 8);
    return u.v;
}

// B-operand (32x16 bf16, K x Ncols) load from "transposed" storage where
// element(kk, col) lives at base[col*ld + kk]. Lane L: col = L&15, b = L>>4;
// holds kk = b*16 .. b*16+15 contiguously -> single 32-byte load.
__device__ __forceinline__ v16bf load_B32x16_t(const bf16_t* base, int ld, int lane) {
    int n = lane & 15, b = lane >> 4;
    return *(const v16bf*)(base + n * ld + b * 16);
}

__device__ __forceinline__ v8f wmma_bf16(v16bf a, v16bf b, v8f c) {
    return __builtin_amdgcn_wmma_f32_16x16x32_bf16(false, a, false, b,
                                                   (short)0, c, false, false);
}

// ---------------------------------------------------------------------------
// Kernel 1: GroupNorm -> x_hat stored transposed [N][C] in bf16
// ---------------------------------------------------------------------------
__global__ void gn_kernel(const float* __restrict__ x,
                          const float* __restrict__ gamma,
                          const float* __restrict__ beta,
                          bf16_t* __restrict__ Xt) {
    __shared__ float s_sum[256];
    __shared__ float s_sq[256];
    int g = blockIdx.x;
    const float* xg = x + (size_t)g * GSIZE;
    float s = 0.f, q = 0.f;
    for (int i = threadIdx.x; i < GSIZE; i += 256) {
        float v = xg[i];
        s += v;
        q += v * v;
    }
    s_sum[threadIdx.x] = s;
    s_sq[threadIdx.x] = q;
    __syncthreads();
    for (int off = 128; off > 0; off >>= 1) {
        if ((int)threadIdx.x < off) {
            s_sum[threadIdx.x] += s_sum[threadIdx.x + off];
            s_sq[threadIdx.x]  += s_sq[threadIdx.x + off];
        }
        __syncthreads();
    }
    float mean = s_sum[0] * (1.0f / (float)GSIZE);
    float var  = s_sq[0] * (1.0f / (float)GSIZE) - mean * mean;
    float rstd = rsqrtf(var + 1e-5f);
    for (int i = threadIdx.x; i < GSIZE; i += 256) {
        int cl = i >> 12;        // channel within group (0..7)
        int n  = i & (NN - 1);
        int c  = g * 8 + cl;
        float v = (xg[i] - mean) * rstd * gamma[c] + beta[c];
        Xt[(size_t)n * CC + c] = (bf16_t)v;
    }
}

// ---------------------------------------------------------------------------
// Kernel 2: weight fp32 -> bf16
// ---------------------------------------------------------------------------
__global__ void cvt_kernel(const float* __restrict__ qkv_w,
                           const float* __restrict__ proj_w,
                           bf16_t* __restrict__ Wq,
                           bf16_t* __restrict__ Wp) {
    int i = blockIdx.x * blockDim.x + threadIdx.x;
    if (i < 3 * CC * CC) Wq[i] = (bf16_t)qkv_w[i];
    if (i < CC * CC)     Wp[i] = (bf16_t)proj_w[i];
}

// ---------------------------------------------------------------------------
// Kernel 3: QKV GEMM  h[768][4096] = Wq[768][256] @ Xhat[256][4096] + b
// One wave -> 16x64 output tile. Scatter into Q,K [head][N][32] bf16
// (Q pre-scaled by hd^-0.5) and V transposed [head][32][N] bf16.
// ---------------------------------------------------------------------------
__global__ void qkv_gemm_kernel(const bf16_t* __restrict__ Wq,
                                const bf16_t* __restrict__ Xt,
                                const float* __restrict__ bias,
                                bf16_t* __restrict__ Q,
                                bf16_t* __restrict__ K,
                                bf16_t* __restrict__ Vt) {
    int lane = threadIdx.x & 31;
    int w = (blockIdx.x * blockDim.x + threadIdx.x) >> 5;   // 0..3071
    int otile = w >> 6;                                     // 0..47
    int n0 = (w & 63) << 6;                                 // 0..4032
    v8f acc[4] = {};
    for (int kk = 0; kk < CC; kk += 32) {
        v16bf a = load_A16x32(Wq + (size_t)otile * 16 * CC + kk, CC, lane);
#pragma unroll
        for (int j = 0; j < 4; ++j) {
            v16bf b = load_B32x16_t(Xt + (size_t)(n0 + 16 * j) * CC + kk, CC, lane);
            acc[j] = wmma_bf16(a, b, acc[j]);
        }
    }
    int hi = lane >> 4, nl = lane & 15;
    int obase = otile * 16;
    int three = obase >> 8;                 // uniform within tile
    const float scale = 0.17677669529663687f;  // 32^-0.5
#pragma unroll
    for (int r = 0; r < 8; ++r) {
        int o = obase + r + 8 * hi;
        float bv = bias[o];
        int head = (o & 255) >> 5;
        int d = o & 31;
#pragma unroll
        for (int j = 0; j < 4; ++j) {
            int n = n0 + 16 * j + nl;
            float val = acc[j][r] + bv;
            if (three == 0) {
                Q[((size_t)head * NN + n) * HD + d] = (bf16_t)(val * scale);
            } else if (three == 1) {
                K[((size_t)head * NN + n) * HD + d] = (bf16_t)val;
            } else {
                Vt[(size_t)head * HD * NN + (size_t)d * NN + n] = (bf16_t)val;
            }
        }
    }
}

// ---------------------------------------------------------------------------
// Kernel 4: flash attention, one wave per (head, 16-query tile).
// ---------------------------------------------------------------------------
__global__ void attn_kernel(const bf16_t* __restrict__ Q,
                            const bf16_t* __restrict__ K,
                            const bf16_t* __restrict__ Vt,
                            bf16_t* __restrict__ Ot) {
    __shared__ __align__(32) bf16_t lds[8][16 * 32];   // per-wave P tile
    int lane = threadIdx.x & 31;
    int wave = threadIdx.x >> 5;
    int w = blockIdx.x * 8 + wave;       // 0..2047
    int head = w >> 8;                   // 0..7
    int q0 = (w & 255) << 4;             // query tile start
    const bf16_t* Qh = Q + (size_t)head * NN * HD;
    const bf16_t* Kh = K + (size_t)head * NN * HD;
    const bf16_t* Vh = Vt + (size_t)head * HD * NN;

    v16bf aq = load_A16x32(Qh + (size_t)q0 * HD, HD, lane);   // Q tile, reused
    v8f o0 = {}, o1 = {};
    float mrow[8], lrow[8];
#pragma unroll
    for (int r = 0; r < 8; ++r) { mrow[r] = -3.0e38f; lrow[r] = 0.f; }

    bf16_t* pl = &lds[wave][0];
    int hi = lane >> 4, nl = lane & 15;

    for (int m0 = 0; m0 < NN; m0 += 32) {
        // scores: S = Q (16x32 over hd) x K^T (32x16)
        v16bf bk0 = load_B32x16_t(Kh + (size_t)m0 * HD, HD, lane);
        v16bf bk1 = load_B32x16_t(Kh + (size_t)(m0 + 16) * HD, HD, lane);
        v8f z = {};
        v8f s0 = wmma_bf16(aq, bk0, z);
        v8f s1 = wmma_bf16(aq, bk1, z);

        // online softmax over this 32-key chunk; row r+8*hi lives in elem r
        // spread across the 16 lanes of this half (xor masks < 16 stay inside).
        float p0[8], p1[8];
#pragma unroll
        for (int r = 0; r < 8; ++r) {
            float t = fmaxf(s0[r], s1[r]);
            t = fmaxf(t, __shfl_xor(t, 1));
            t = fmaxf(t, __shfl_xor(t, 2));
            t = fmaxf(t, __shfl_xor(t, 4));
            t = fmaxf(t, __shfl_xor(t, 8));
            float mn = fmaxf(mrow[r], t);
            float alpha = __expf(mrow[r] - mn);
            mrow[r] = mn;
            float e0 = __expf(s0[r] - mn);
            float e1 = __expf(s1[r] - mn);
            float u = e0 + e1;
            u += __shfl_xor(u, 1);
            u += __shfl_xor(u, 2);
            u += __shfl_xor(u, 4);
            u += __shfl_xor(u, 8);
            lrow[r] = lrow[r] * alpha + u;
            o0[r] = o0[r] * alpha;
            o1[r] = o1[r] * alpha;
            p0[r] = e0;
            p1[r] = e1;
        }

        // re-stripe P from C-layout to A-layout via per-wave LDS tile
        // (DS ops are in-order within a wave)
#pragma unroll
        for (int r = 0; r < 8; ++r) {
            int row = r + 8 * hi;
            pl[row * 32 + nl]      = (bf16_t)p0[r];
            pl[row * 32 + nl + 16] = (bf16_t)p1[r];
        }
        v16bf ap = load_A16x32(pl, 32, lane);

        // O += P (16x32 over keys) x V (32x16 over hd halves)
        v16bf bv0 = load_B32x16_t(Vh + m0, NN, lane);                    // d 0..15
        v16bf bv1 = load_B32x16_t(Vh + (size_t)16 * NN + m0, NN, lane);  // d 16..31
        o0 = wmma_bf16(ap, bv0, o0);
        o1 = wmma_bf16(ap, bv1, o1);
    }

    // normalize and store transposed [N][C] (c = head*32 + d) for proj GEMM
#pragma unroll
    for (int r = 0; r < 8; ++r) {
        float inv = 1.0f / lrow[r];
        int nq = q0 + r + 8 * hi;
        Ot[(size_t)nq * CC + head * HD + nl]      = (bf16_t)(o0[r] * inv);
        Ot[(size_t)nq * CC + head * HD + nl + 16] = (bf16_t)(o1[r] * inv);
    }
}

// ---------------------------------------------------------------------------
// Kernel 5: proj GEMM + bias + residual -> fp32 output [C][N]
// ---------------------------------------------------------------------------
__global__ void proj_gemm_kernel(const bf16_t* __restrict__ Wp,
                                 const bf16_t* __restrict__ At,
                                 const float* __restrict__ bias,
                                 const float* __restrict__ x,
                                 float* __restrict__ out) {
    int lane = threadIdx.x & 31;
    int w = (blockIdx.x * blockDim.x + threadIdx.x) >> 5;   // 0..1023
    int otile = w >> 6;                                     // 0..15
    int n0 = (w & 63) << 6;
    v8f acc[4] = {};
    for (int kk = 0; kk < CC; kk += 32) {
        v16bf a = load_A16x32(Wp + (size_t)otile * 16 * CC + kk, CC, lane);
#pragma unroll
        for (int j = 0; j < 4; ++j) {
            v16bf b = load_B32x16_t(At + (size_t)(n0 + 16 * j) * CC + kk, CC, lane);
            acc[j] = wmma_bf16(a, b, acc[j]);
        }
    }
    int hi = lane >> 4, nl = lane & 15;
#pragma unroll
    for (int r = 0; r < 8; ++r) {
        int o = otile * 16 + r + 8 * hi;
        float bv = bias[o];
#pragma unroll
        for (int j = 0; j < 4; ++j) {
            int n = n0 + 16 * j + nl;
            size_t idx = (size_t)o * NN + n;
            out[idx] = x[idx] + acc[j][r] + bv;
        }
    }
}

// ---------------------------------------------------------------------------
extern "C" void kernel_launch(void* const* d_in, const int* in_sizes, int n_in,
                              void* d_out, int out_size, void* d_ws, size_t ws_size,
                              hipStream_t stream) {
    const float* x        = (const float*)d_in[0];
    const float* gn_gamma = (const float*)d_in[1];
    const float* gn_beta  = (const float*)d_in[2];
    const float* qkv_w    = (const float*)d_in[3];
    const float* qkv_b    = (const float*)d_in[4];
    const float* proj_w   = (const float*)d_in[5];
    const float* proj_b   = (const float*)d_in[6];
    float* out = (float*)d_out;

    // workspace carve (~10.5 MB total, all offsets 256B-aligned)
    char* p = (char*)d_ws;
    bf16_t* Xt  = (bf16_t*)p; p += (size_t)NN * CC * 2;          // 2 MB
    bf16_t* Wq  = (bf16_t*)p; p += (size_t)3 * CC * CC * 2;      // 384 KB
    bf16_t* Wp  = (bf16_t*)p; p += (size_t)CC * CC * 2;          // 128 KB
    bf16_t* Qb  = (bf16_t*)p; p += (size_t)HEADS * NN * HD * 2;  // 2 MB
    bf16_t* Kb  = (bf16_t*)p; p += (size_t)HEADS * NN * HD * 2;  // 2 MB
    bf16_t* Vtb = (bf16_t*)p; p += (size_t)HEADS * NN * HD * 2;  // 2 MB
    bf16_t* Ot  = (bf16_t*)p; p += (size_t)NN * CC * 2;          // 2 MB

    gn_kernel<<<GROUPS, 256, 0, stream>>>(x, gn_gamma, gn_beta, Xt);
    cvt_kernel<<<768, 256, 0, stream>>>(qkv_w, proj_w, Wq, Wp);
    // 48 o-tiles * 64 n-groups = 3072 waves / 8 waves-per-block
    qkv_gemm_kernel<<<384, 256, 0, stream>>>(Wq, Xt, qkv_b, Qb, Kb, Vtb);
    // 8 heads * 256 q-tiles = 2048 waves / 8
    attn_kernel<<<256, 256, 0, stream>>>(Qb, Kb, Vtb, Ot);
    // 16 o-tiles * 64 n-groups = 1024 waves / 8
    proj_gemm_kernel<<<128, 256, 0, stream>>>(Wp, Ot, proj_b, x, out);
}
